// HandshakingKernel_30459908063855
// MI455X (gfx1250) — compile-verified, matrix operand order checked by reading, not run on hardware
//
#include <hip/hip_runtime.h>
#include <stdint.h>

#define S_   512
#define H_   768
#define B_   2
#define VF_  64
#define NV_  30752
#define EPS_ 1e-12f

typedef __attribute__((ext_vector_type(16))) _Float16 v16h;
typedef __attribute__((ext_vector_type(8)))  _Float16 v8h;
typedef __attribute__((ext_vector_type(8)))  float    v8f;

// Load one 16x32 f16 WMMA fragment for this lane from a row-major matrix row.
// CDNA5 ISA layout (16-bit A/B 16x32):
//   lanes 0-15 : VGPRs 0-3 hold K=kb+0..7,  VGPRs 4-7 hold K=kb+16..23
//   lanes 16-31: VGPRs 0-3 hold K=kb+8..15, VGPRs 4-7 hold K=kb+24..31
static __device__ __forceinline__ v16h frag_load(const _Float16* __restrict__ rowptr,
                                                 int kb, int hi) {
  const v8h* p0 = reinterpret_cast<const v8h*>(rowptr + kb + hi * 8);
  const v8h* p1 = reinterpret_cast<const v8h*>(rowptr + kb + 16 + hi * 8);
  v8h lo = *p0;
  v8h hh = *p1;
  v16h r;
#pragma unroll
  for (int j = 0; j < 8; ++j) { r[j] = lo[j]; r[j + 8] = hh[j]; }
  return r;
}

static __device__ __forceinline__ float sigmoidf_(float x) {
  return 1.0f / (1.0f + __expf(-x));
}

// Async-copy `nbytes16` 16-byte chunks from global `src` into LDS at `dst`
// (both contiguous) using GLOBAL_LOAD_ASYNC_TO_LDS_B128 (ASYNCcnt-tracked).
static __device__ __forceinline__ void async_copy_tile(const _Float16* __restrict__ src,
                                                       _Float16* dst, int nchunks, int tid) {
  uint32_t lds_base = (uint32_t)(uintptr_t)dst;   // low 32 bits = LDS byte address
  const char* gbase = (const char*)src;
  for (int c = tid; c < nchunks; c += 256) {
    uint32_t lo = lds_base + (uint32_t)c * 16u;
    const char* gp = gbase + (size_t)c * 16u;
    asm volatile("global_load_async_to_lds_b128 %0, %1, off"
                 :: "v"(lo), "v"(gp) : "memory");
  }
  asm volatile("s_wait_asynccnt 0x0" ::: "memory");
}

// ---------------- Kernel 1: prefix/suffix scans -> cs0, bf_pool, af_pool ----
__global__ void scan_kernel(const float* __restrict__ h, const float* __restrict__ lamtha,
                            float* __restrict__ cs0, float* __restrict__ bf_pool,
                            float* __restrict__ af_pool) {
  int idx = blockIdx.x * blockDim.x + threadIdx.x;  // 0 .. B*H-1
  if (idx >= B_ * H_) return;
  int b = idx / H_, c = idx % H_;
  float lam = lamtha[c];
  const float* hb = h + (size_t)b * S_ * H_ + c;
  float* csb = cs0 + (size_t)b * (S_ + 1) * H_ + c;
  float* bfb = bf_pool + (size_t)b * S_ * H_ + c;
  float* afb = af_pool + (size_t)b * S_ * H_ + c;

  float run = 0.f, pmax = 0.f;
  csb[0] = 0.f;
  for (int s = 0; s < S_; ++s) {
    float x = hb[(size_t)s * H_];
    if (s == 0) {
      bfb[0] = 0.f;
    } else {
      float bfm = run / (float)s;                       // mean of h[0..s-1]
      bfb[(size_t)s * H_] = lam * bfm + (1.f - lam) * pmax;
    }
    run += x;
    pmax = (s == 0) ? x : fmaxf(pmax, x);
    csb[(size_t)(s + 1) * H_] = run;
  }
  float ssum = 0.f, smax = 0.f;
  for (int s = S_ - 1; s >= 0; --s) {
    float x = hb[(size_t)s * H_];
    if (s == S_ - 1) {
      afb[(size_t)s * H_] = 0.f;
    } else {
      float afm = ssum / (float)(S_ - 1 - s);           // mean of h[s+1..S-1]
      afb[(size_t)s * H_] = lam * afm + (1.f - lam) * smax;
    }
    ssum += x;
    smax = (s == S_ - 1) ? x : fmaxf(smax, x);
  }
}

// ---------------- Kernel 2: per-token row stats -> xn = (h-mean)/(var+eps)^2,
//                  plus f16 copy of h (GEMM1 A matrix) -----------------------
__global__ void rowstat_kernel(const float* __restrict__ h, float* __restrict__ xn,
                               _Float16* __restrict__ h16) {
  int row = blockIdx.x;        // 0 .. B*S-1
  int tid = threadIdx.x;       // 256
  const float* hp = h + (size_t)row * H_;
  float v[3];
  float s1 = 0.f;
#pragma unroll
  for (int k = 0; k < 3; ++k) { v[k] = hp[tid + 256 * k]; s1 += v[k]; }
  __shared__ float red[256];
  red[tid] = s1; __syncthreads();
  for (int off = 128; off >= 1; off >>= 1) {
    if (tid < off) red[tid] += red[tid + off];
    __syncthreads();
  }
  float mean = red[0] * (1.0f / (float)H_);
  __syncthreads();
  float d[3];
  float s2 = 0.f;
#pragma unroll
  for (int k = 0; k < 3; ++k) { d[k] = v[k] - mean; s2 += d[k] * d[k]; }
  red[tid] = s2; __syncthreads();
  for (int off = 128; off >= 1; off >>= 1) {
    if (tid < off) red[tid] += red[tid + off];
    __syncthreads();
  }
  float var = red[0] * (1.0f / (float)H_);
  float inv = 1.0f / ((var + EPS_) * (var + EPS_));     // reference: std=(var+eps)^2
  float* xp = xn + (size_t)row * H_;
  _Float16* hq = h16 + (size_t)row * H_;
#pragma unroll
  for (int k = 0; k < 3; ++k) {
    xp[tid + 256 * k] = d[k] * inv;
    hq[tid + 256 * k] = (_Float16)v[k];
  }
}

// ---------------- Kernel 3: f32 -> f16 weight conversion --------------------
__global__ void f2h_kernel(const float* __restrict__ src, _Float16* __restrict__ dst, int n) {
  int i = blockIdx.x * 256 + threadIdx.x;
  if (i < n) dst[i] = (_Float16)src[i];
}

// ---------------- Kernel 4: GEMM1 (M=B*S=1024) -> beta1_act/gamma1_act ------
// block = 8 waves; covers 16 rows x 384 cols (half of N) for both weights.
// wave owns 3 N-tiles x {Wb,Wg}. A tile staged in LDS via async copy.
__global__ void __launch_bounds__(256) gemm1_kernel(
    const _Float16* __restrict__ A, const _Float16* __restrict__ Wb,
    const _Float16* __restrict__ Wg, const float* __restrict__ beta_p,
    const float* __restrict__ gamma_p, float* __restrict__ bact, float* __restrict__ gact) {
  __shared__ _Float16 ldsA[16 * H_];                 // 24 KB
  int r0 = (blockIdx.x >> 1) * 16;
  int half = blockIdx.x & 1;
  int tid = threadIdx.x;

  async_copy_tile(A + (size_t)r0 * H_, ldsA, 16 * H_ * 2 / 16, tid);
  __syncthreads();

  int w = tid >> 5, l = tid & 31;
  int hi = l >> 4, lm = l & 15;
  const _Float16* arow = &ldsA[lm * H_];
  int nbase = half * 384 + w * 48 + lm;
  const _Float16* pB[3];
  const _Float16* pG[3];
#pragma unroll
  for (int t = 0; t < 3; ++t) {
    pB[t] = Wb + (size_t)(nbase + t * 16) * H_;
    pG[t] = Wg + (size_t)(nbase + t * 16) * H_;
  }
  v8f zero = {0.f, 0.f, 0.f, 0.f, 0.f, 0.f, 0.f, 0.f};
  v8f accB[3], accG[3];
#pragma unroll
  for (int t = 0; t < 3; ++t) { accB[t] = zero; accG[t] = zero; }

#pragma unroll
  for (int kb = 0; kb < H_; kb += 32) {
    v16h af = frag_load(arow, kb, hi);               // ds_load from LDS
    v16h b0 = frag_load(pB[0], kb, hi);
    v16h b1 = frag_load(pB[1], kb, hi);
    v16h b2 = frag_load(pB[2], kb, hi);
    v16h g0 = frag_load(pG[0], kb, hi);
    v16h g1 = frag_load(pG[1], kb, hi);
    v16h g2 = frag_load(pG[2], kb, hi);
    accB[0] = __builtin_amdgcn_wmma_f32_16x16x32_f16(false, af, false, b0, (short)0, accB[0], false, false);
    accB[1] = __builtin_amdgcn_wmma_f32_16x16x32_f16(false, af, false, b1, (short)0, accB[1], false, false);
    accB[2] = __builtin_amdgcn_wmma_f32_16x16x32_f16(false, af, false, b2, (short)0, accB[2], false, false);
    accG[0] = __builtin_amdgcn_wmma_f32_16x16x32_f16(false, af, false, g0, (short)0, accG[0], false, false);
    accG[1] = __builtin_amdgcn_wmma_f32_16x16x32_f16(false, af, false, g1, (short)0, accG[1], false, false);
    accG[2] = __builtin_amdgcn_wmma_f32_16x16x32_f16(false, af, false, g2, (short)0, accG[2], false, false);
  }
#pragma unroll
  for (int t = 0; t < 3; ++t) {
    int ncol = nbase + t * 16;
    float bp = beta_p[ncol], gp = gamma_p[ncol];
#pragma unroll
    for (int r = 0; r < 8; ++r) {
      int m = r + hi * 8;
      size_t o = (size_t)(r0 + m) * H_ + ncol;
      bact[o] = sigmoidf_(accB[t][r]) + bp;
      gact[o] = sigmoidf_(accG[t][r]) + gp;
    }
  }
}

// ---------------- Kernel 5: per-window context -> ctx16 (f16, GEMM row order)
__global__ void context_kernel(const float* __restrict__ h, const float* __restrict__ lamtha,
                               const float* __restrict__ cs0, const float* __restrict__ bf_pool,
                               const float* __restrict__ af_pool, _Float16* __restrict__ ctx16) {
  int bs = blockIdx.x;                 // 0 .. B*S-1
  int b = bs / S_, s = bs % S_;
  int tid = threadIdx.x;
  const float* hb = h + (size_t)b * S_ * H_;
  const float* csb = cs0 + (size_t)b * (S_ + 1) * H_;
  float lam[3], bfv[3], css[3], runmax[3];
#pragma unroll
  for (int k = 0; k < 3; ++k) {
    int c = tid + 256 * k;
    lam[k] = lamtha[c];
    bfv[k] = bf_pool[(size_t)bs * H_ + c];
    css[k] = csb[(size_t)s * H_ + c];
    runmax[k] = 0.f;
  }
  _Float16* cp = ctx16 + (size_t)bs * VF_ * H_;
  for (int i = 0; i < VF_; ++i) {
    int t = min(s + i, S_ - 1);                 // clamped index, matches reference
    int tp1 = min(s + i + 1, S_);
    float inv = 1.0f / (float)(i + 1);
#pragma unroll
    for (int k = 0; k < 3; ++k) {
      int c = tid + 256 * k;
      float hv = hb[(size_t)t * H_ + c];
      runmax[k] = (i == 0) ? hv : fmaxf(runmax[k], hv);      // ent_max (cummax)
      float emean = (csb[(size_t)tp1 * H_ + c] - css[k]) * inv;
      float ep = lam[k] * emean + (1.f - lam[k]) * runmax[k];
      float afv = af_pool[(size_t)(b * S_ + t) * H_ + c];
      float ctx = (ep + bfv[k] + afv) * (1.0f / 3.0f);
      cp[(size_t)i * H_ + c] = (_Float16)ctx;
    }
  }
}

// ---------------- Kernel 6: fused GEMM2 (beta2/gamma2) + cln2 + scatter -----
// block = 8 waves; covers 16 window-rows x 384 cols (half of N), both weights.
__global__ void __launch_bounds__(256) gemm2_fused_kernel(
    const _Float16* __restrict__ ctx16, const _Float16* __restrict__ Wb,
    const _Float16* __restrict__ Wg, const float* __restrict__ beta_p,
    const float* __restrict__ gamma_p, const float* __restrict__ xn,
    const float* __restrict__ b1act, const float* __restrict__ g1act,
    float* __restrict__ out) {
  const int LSTR = 772;                       // padded row stride (bank spread)
  __shared__ float lds1[16 * LSTR];           // out1 rows (16 x 768), ~48 KB
  __shared__ _Float16 ldsA[16 * H_];          // A tile (ctx16 rows), 24 KB
  __shared__ float rmean[16], rinv[16];

  int r0 = (blockIdx.x >> 1) * 16;            // global window-row base
  int half = blockIdx.x & 1;
  int b = r0 / (S_ * VF_);
  int rem = r0 % (S_ * VF_);
  int s = rem / VF_;
  int i0 = rem % VF_;                         // multiple of 16 (VF=64)
  int tid = threadIdx.x;

  // Kick off async A-tile copy (24 KB contiguous) into LDS.
  async_copy_tile(ctx16 + (size_t)r0 * H_, ldsA, 16 * H_ * 2 / 16, tid);

  // Stage out1[m][c] = xn[b, min(s+i0+m, S-1)][c] * gamma1_act[b,s][c] + beta1_act[b,s][c]
  const float* g1 = g1act + (size_t)(b * S_ + s) * H_;
  const float* b1 = b1act + (size_t)(b * S_ + s) * H_;
  for (int e = tid; e < 16 * H_; e += 256) {
    int row = e / H_, c = e - row * H_;
    int t = min(s + i0 + row, S_ - 1);
    lds1[row * LSTR + c] = xn[(size_t)(b * S_ + t) * H_ + c] * g1[c] + b1[c];
  }
  __syncthreads();

  int w = tid >> 5, l = tid & 31;
  // wave w computes cln2 stats for rows 2w, 2w+1 via shuffle reduction
#pragma unroll
  for (int rr = 0; rr < 2; ++rr) {
    int row = w * 2 + rr;
    float s1 = 0.f;
    for (int c = l; c < H_; c += 32) s1 += lds1[row * LSTR + c];
#pragma unroll
    for (int off = 16; off >= 1; off >>= 1) s1 += __shfl_xor(s1, off);
    float mean = s1 * (1.0f / (float)H_);
    float s2 = 0.f;
    for (int c = l; c < H_; c += 32) { float d = lds1[row * LSTR + c] - mean; s2 += d * d; }
#pragma unroll
    for (int off = 16; off >= 1; off >>= 1) s2 += __shfl_xor(s2, off);
    float var = s2 * (1.0f / (float)H_);
    if (l == 0) { rmean[row] = mean; rinv[row] = 1.0f / ((var + EPS_) * (var + EPS_)); }
  }
  __syncthreads();

  int hi = l >> 4, lm = l & 15;
  const _Float16* arow = &ldsA[lm * H_];
  int nbase = half * 384 + w * 48 + lm;
  const _Float16* pB[3];
  const _Float16* pG[3];
#pragma unroll
  for (int t = 0; t < 3; ++t) {
    pB[t] = Wb + (size_t)(nbase + t * 16) * H_;
    pG[t] = Wg + (size_t)(nbase + t * 16) * H_;
  }
  v8f zero = {0.f, 0.f, 0.f, 0.f, 0.f, 0.f, 0.f, 0.f};
  v8f accB[3], accG[3];
#pragma unroll
  for (int t = 0; t < 3; ++t) { accB[t] = zero; accG[t] = zero; }

#pragma unroll
  for (int kb = 0; kb < H_; kb += 32) {
    v16h af = frag_load(arow, kb, hi);               // ds_load from LDS
    v16h b0 = frag_load(pB[0], kb, hi);
    v16h b1f = frag_load(pB[1], kb, hi);
    v16h b2 = frag_load(pB[2], kb, hi);
    v16h g0 = frag_load(pG[0], kb, hi);
    v16h g1f = frag_load(pG[1], kb, hi);
    v16h g2 = frag_load(pG[2], kb, hi);
    accB[0] = __builtin_amdgcn_wmma_f32_16x16x32_f16(false, af, false, b0, (short)0, accB[0], false, false);
    accB[1] = __builtin_amdgcn_wmma_f32_16x16x32_f16(false, af, false, b1f, (short)0, accB[1], false, false);
    accB[2] = __builtin_amdgcn_wmma_f32_16x16x32_f16(false, af, false, b2, (short)0, accB[2], false, false);
    accG[0] = __builtin_amdgcn_wmma_f32_16x16x32_f16(false, af, false, g0, (short)0, accG[0], false, false);
    accG[1] = __builtin_amdgcn_wmma_f32_16x16x32_f16(false, af, false, g1f, (short)0, accG[1], false, false);
    accG[2] = __builtin_amdgcn_wmma_f32_16x16x32_f16(false, af, false, g2, (short)0, accG[2], false, false);
  }

  // Epilogue: sigmoid + params, cln2 normalize, scatter valid rows only.
  int valid_n = S_ - s;
  int obase = (s <= S_ - VF_) ? s * VF_ : (NV_ - (S_ - s) * (S_ - s + 1) / 2);
  float* outb = out + (size_t)b * NV_ * H_;
#pragma unroll
  for (int t = 0; t < 3; ++t) {
    int ncol = nbase + t * 16;
    float bp = beta_p[ncol], gp = gamma_p[ncol];
#pragma unroll
    for (int r = 0; r < 8; ++r) {
      int m = r + hi * 8;
      int i = i0 + m;
      float beta2v = sigmoidf_(accB[t][r]) + bp;
      float gamma2v = sigmoidf_(accG[t][r]) + gp;
      float o = lds1[m * LSTR + ncol] - rmean[m];
      float val = o * rinv[m] * gamma2v + beta2v;
      if (i < valid_n) outb[(size_t)(obase + i) * H_ + ncol] = val;
    }
  }
}

extern "C" void kernel_launch(void* const* d_in, const int* in_sizes, int n_in,
                              void* d_out, int out_size, void* d_ws, size_t ws_size,
                              hipStream_t stream) {
  (void)in_sizes; (void)n_in; (void)out_size; (void)ws_size;
  const float* h      = (const float*)d_in[0];
  const float* lamtha = (const float*)d_in[1];
  const float* beta1  = (const float*)d_in[2];
  const float* gamma1 = (const float*)d_in[3];
  const float* Wb1    = (const float*)d_in[4];
  const float* Wg1    = (const float*)d_in[5];
  const float* beta2  = (const float*)d_in[6];
  const float* gamma2 = (const float*)d_in[7];
  const float* Wb2    = (const float*)d_in[8];
  const float* Wg2    = (const float*)d_in[9];
  float* out = (float*)d_out;

  char* wp = (char*)d_ws;
  auto carve = [&](size_t bytes) -> void* {
    void* p = (void*)wp;
    wp += (bytes + 255) & ~(size_t)255;
    return p;
  };
  float*    cs0   = (float*)carve((size_t)B_ * (S_ + 1) * H_ * 4);
  float*    bfp   = (float*)carve((size_t)B_ * S_ * H_ * 4);
  float*    afp   = (float*)carve((size_t)B_ * S_ * H_ * 4);
  float*    xn    = (float*)carve((size_t)B_ * S_ * H_ * 4);
  _Float16* h16   = (_Float16*)carve((size_t)B_ * S_ * H_ * 2);
  _Float16* w1b   = (_Float16*)carve((size_t)H_ * H_ * 2);
  _Float16* w1g   = (_Float16*)carve((size_t)H_ * H_ * 2);
  _Float16* w2b   = (_Float16*)carve((size_t)H_ * H_ * 2);
  _Float16* w2g   = (_Float16*)carve((size_t)H_ * H_ * 2);
  float*    b1act = (float*)carve((size_t)B_ * S_ * H_ * 4);
  float*    g1act = (float*)carve((size_t)B_ * S_ * H_ * 4);
  _Float16* ctx16 = (_Float16*)carve((size_t)B_ * S_ * VF_ * H_ * 2);

  scan_kernel<<<(B_ * H_ + 255) / 256, 256, 0, stream>>>(h, lamtha, cs0, bfp, afp);
  rowstat_kernel<<<B_ * S_, 256, 0, stream>>>(h, xn, h16);
  int nw = H_ * H_;
  f2h_kernel<<<(nw + 255) / 256, 256, 0, stream>>>(Wb1, w1b, nw);
  f2h_kernel<<<(nw + 255) / 256, 256, 0, stream>>>(Wg1, w1g, nw);
  f2h_kernel<<<(nw + 255) / 256, 256, 0, stream>>>(Wb2, w2b, nw);
  f2h_kernel<<<(nw + 255) / 256, 256, 0, stream>>>(Wg2, w2g, nw);
  gemm1_kernel<<<(B_ * S_ / 16) * 2, 256, 0, stream>>>(h16, w1b, w1g, beta1, gamma1, b1act, g1act);
  context_kernel<<<B_ * S_, 256, 0, stream>>>(h, lamtha, cs0, bfp, afp, ctx16);
  gemm2_fused_kernel<<<(B_ * S_ * VF_ / 16) * 2, 256, 0, stream>>>(ctx16, w2b, w2g, beta2, gamma2,
                                                                   xn, b1act, g1act, out);
}
